// RetrievalModel_70849780515081
// MI455X (gfx1250) — matrix operand order repaired and checked by reading, not compiled, not run on hardware
//
#include <hip/hip_runtime.h>
#include <hip/hip_bf16.h>
#include <math.h>

typedef __attribute__((ext_vector_type(16))) __bf16 v16bf;
typedef __attribute__((ext_vector_type(8)))  __bf16 v8bf;
typedef __attribute__((ext_vector_type(8)))  float  v8f;
typedef __attribute__((ext_vector_type(4)))  float  v4f;
typedef __attribute__((ext_vector_type(2)))  float  v2f;

#define B_ROWS 2048
#define N_DB   100000
#define DIM    1024
#define ES_N   101
#define H1_N   256
#define H2_N   128
#define TOPK   90
#define NSPLIT 4
#define MROWS  32     // q rows per block in the sims kernel (2 x 16-row A tiles)
#define FK     1128   // 1125 padded to multiple of 4 for 16x16x4 f32 WMMA

// ---------------------------------------------------------------------------
// Kernel 1: fused L2-normalize + f32 -> bf16 downconvert (one wave per row).
// ---------------------------------------------------------------------------
__global__ void __launch_bounds__(256) l2norm_to_bf16(
    const float* __restrict__ src, __bf16* __restrict__ dst, int rows)
{
  int row  = blockIdx.x * 8 + (threadIdx.x >> 5);
  int lane = threadIdx.x & 31;
  if (row >= rows) return;
  const float* s = src + (size_t)row * DIM;
  float x[32];
  float ss = 0.f;
  #pragma unroll
  for (int i = 0; i < 32; ++i) { x[i] = s[lane + 32 * i]; ss += x[i] * x[i]; }
  #pragma unroll
  for (int m = 16; m >= 1; m >>= 1) ss += __shfl_xor(ss, m, 32);
  float inv = 1.f / fmaxf(sqrtf(ss), 1e-12f);
  __bf16* d = dst + (size_t)row * DIM;
  #pragma unroll
  for (int i = 0; i < 32; ++i) d[lane + 32 * i] = (__bf16)(x[i] * inv);
}

// ---------------------------------------------------------------------------
// Kernel 2: fused sims GEMM (bf16 WMMA) + per-row running top-90.
// Block = 32 q-rows (two 16-row A tiles sharing each B fragment -> 2x
// arithmetic intensity per db byte) x one N-slice; 8 waves x 16 db cols.
// q tile staged into LDS via async global->LDS loads (ASYNCcnt path).
// Top-k: threshold-filtered candidate queue (LDS atomics) -> serial insert
// touches only survivors.
// ---------------------------------------------------------------------------
__global__ void __launch_bounds__(256) sims_topk_kernel(
    const __bf16* __restrict__ qn, const __bf16* __restrict__ dbn,
    float* __restrict__ pvals, int* __restrict__ pidx)
{
  __shared__ __align__(16) __bf16 qs[MROWS * DIM];   // 64 KB q tile
  __shared__ float qval[MROWS * 128];                // 16 KB queue values
  __shared__ int   qcol[MROWS * 128];                // 16 KB queue columns
  __shared__ int   qcnt[MROWS];
  __shared__ __align__(16) float thr[MROWS];         // per-row 90th-best
  __shared__ float tv[MROWS * TOPK];                 // running top-90 values
  __shared__ int   ti[MROWS * TOPK];                 // running top-90 indices

  const int tid   = threadIdx.x;
  const int wave  = tid >> 5;
  const int lane  = tid & 31;
  const int b0    = blockIdx.x * MROWS;
  const int slice = blockIdx.y;

  // --- async-stage 32x1024 bf16 q tile into LDS (256 B per thread) ---
  {
    unsigned lds_base = (unsigned)(size_t)(void*)&qs[0];
    unsigned long long g =
        (unsigned long long)(const void*)(qn + (size_t)b0 * DIM);
    #pragma unroll
    for (int i = 0; i < 16; ++i) {
      unsigned loff = (unsigned)tid * 256u + (unsigned)i * 16u;
      unsigned ldsa = lds_base + loff;
      unsigned long long ga = g + loff;
      asm volatile("global_load_async_to_lds_b128 %0, %1, off"
                   :: "v"(ldsa), "v"(ga) : "memory");
    }
    asm volatile("s_wait_asynccnt 0" ::: "memory");
  }
  if (tid < MROWS) { qcnt[tid] = 0; thr[tid] = -3.4e38f; }
  __syncthreads();

  const int n_per = (N_DB + NSPLIT - 1) / NSPLIT;
  const int n0 = slice * n_per;
  const int n1 = min(N_DB, n0 + n_per);

  // per-row top-k state (only tid<MROWS use these)
  int   kcnt = 0, kminp = 0;
  float kminv = -3.4e38f;

  const int mrow  = lane & 15;
  const int kbase = (lane < 16) ? 0 : 8;   // A-frag K interleave (ISA 7.12.2)
  const int kbB   = (lane < 16) ? 0 : 16;  // unused; B-frag uses kbase interleave
  const int rbase = (lane < 16) ? 0 : 8;   // C-frag row base

  for (int base = n0; base < n1; base += 128) {
    int colg = base + wave * 16 + (lane & 15);
    int nc = (colg < n1) ? colg : (n1 - 1);           // clamp; dropped later
    const __bf16* brow = dbn + (size_t)nc * DIM;

    v8f c0 = {}, c1 = {};
    #pragma unroll 4
    for (int kc = 0; kc < DIM; kc += 32) {
      const __bf16* ap0 = &qs[mrow * DIM + kc + kbase];
      const __bf16* ap1 = &qs[(16 + mrow) * DIM + kc + kbase];
      v8bf a0lo = *(const v8bf*)(ap0);
      v8bf a0hi = *(const v8bf*)(ap0 + 16);
      v8bf a1lo = *(const v8bf*)(ap1);
      v8bf a1hi = *(const v8bf*)(ap1 + 16);
      v16bf a0, a1, b;
      v8bf blo = *(const v8bf*)(brow + kc + kbase);
      v8bf bhi = *(const v8bf*)(brow + kc + kbase + 16);
      #pragma unroll
      for (int e = 0; e < 8; ++e) {
        a0[e] = a0lo[e]; a0[e + 8] = a0hi[e];
        b[e] = blo[e]; b[e + 8] = bhi[e];
        a1[e] = a1lo[e]; a1[e + 8] = a1hi[e];
      }
      c0 = __builtin_amdgcn_wmma_f32_16x16x32_bf16(
               false, a0, false, b, (short)0, c0, false, false);
      c1 = __builtin_amdgcn_wmma_f32_16x16x32_bf16(
               false, a1, false, b, (short)0, c1, false, false);
    }

    // parallel threshold filter straight from the accumulators
    if (colg < n1) {
      float tloc[16];
      *(v4f*)&tloc[0]  = *(const v4f*)&thr[rbase];
      *(v4f*)&tloc[4]  = *(const v4f*)&thr[rbase + 4];
      *(v4f*)&tloc[8]  = *(const v4f*)&thr[16 + rbase];
      *(v4f*)&tloc[12] = *(const v4f*)&thr[16 + rbase + 4];
      #pragma unroll
      for (int r = 0; r < 8; ++r) {
        if (c0[r] > tloc[r]) {
          int m = rbase + r;
          int pos = atomicAdd(&qcnt[m], 1);   // ds_add_rtn_u32; pos < 128 always
          qval[m * 128 + pos] = c0[r];
          qcol[m * 128 + pos] = colg;
        }
      }
      #pragma unroll
      for (int r = 0; r < 8; ++r) {
        if (c1[r] > tloc[8 + r]) {
          int m = 16 + rbase + r;
          int pos = atomicAdd(&qcnt[m], 1);
          qval[m * 128 + pos] = c1[r];
          qcol[m * 128 + pos] = colg;
        }
      }
    }
    __syncthreads();

    // owner thread per q-row inserts only the survivors
    if (tid < MROWS) {
      const int m = tid;
      const int nq = qcnt[m];
      float* v = &tv[m * TOPK]; int* ix = &ti[m * TOPK];
      for (int qq = 0; qq < nq; ++qq) {
        float s  = qval[m * 128 + qq];
        int gcol = qcol[m * 128 + qq];
        if (kcnt < TOPK) {
          v[kcnt] = s; ix[kcnt] = gcol; ++kcnt;
          if (kcnt == TOPK) {
            kminv = v[0]; kminp = 0;
            for (int j = 1; j < TOPK; ++j)
              if (v[j] < kminv) { kminv = v[j]; kminp = j; }
          }
        } else if (s > kminv) {
          v[kminp] = s; ix[kminp] = gcol;
          kminv = v[0]; kminp = 0;
          for (int j = 1; j < TOPK; ++j)
            if (v[j] < kminv) { kminv = v[j]; kminp = j; }
        }
      }
      qcnt[m] = 0;
      thr[m] = (kcnt == TOPK) ? kminv : -3.4e38f;
    }
    __syncthreads();
  }

  if (tid < MROWS) {
    int b = b0 + tid;
    for (int k = 0; k < TOPK; ++k) {
      pvals[((size_t)b * NSPLIT + slice) * TOPK + k] = tv[tid * TOPK + k];
      pidx [((size_t)b * NSPLIT + slice) * TOPK + k] = ti[tid * TOPK + k];
    }
  }
}

// ---------------------------------------------------------------------------
// Kernel 3: merge 4 partial top-90s -> global top-90, softmax(T=0.04),
// weighted gather of es_db restricted to the 101 sp_idx columns.
// ---------------------------------------------------------------------------
__global__ void __launch_bounds__(128) merge_softmax_retrieve(
    const float* __restrict__ pvals, const int* __restrict__ pidx,
    const float* __restrict__ es_db, const int* __restrict__ sp_idx,
    float* __restrict__ es_ret, float* __restrict__ out_ret)
{
  __shared__ float cv[NSPLIT * TOPK];
  __shared__ int   ci[NSPLIT * TOPK];
  __shared__ float rv[128];
  __shared__ int   rs[128];
  __shared__ float selv[TOPK];
  __shared__ int   seli[TOPK];
  __shared__ float w[TOPK];
  __shared__ float wsum;

  const int tid = threadIdx.x;
  const int b   = blockIdx.x;
  const int M   = NSPLIT * TOPK;  // 360 candidates

  for (int j = tid; j < M; j += 128) {
    cv[j] = pvals[(size_t)b * M + j];
    ci[j] = pidx [(size_t)b * M + j];
  }
  __syncthreads();

  for (int k = 0; k < TOPK; ++k) {
    float best = -3.4e38f; int bs = 0;
    for (int j = tid; j < M; j += 128)
      if (cv[j] > best) { best = cv[j]; bs = j; }
    rv[tid] = best; rs[tid] = bs;
    __syncthreads();
    for (int s = 64; s > 0; s >>= 1) {
      if (tid < s && rv[tid + s] > rv[tid]) { rv[tid] = rv[tid + s]; rs[tid] = rs[tid + s]; }
      __syncthreads();
    }
    if (tid == 0) { selv[k] = rv[0]; seli[k] = ci[rs[0]]; cv[rs[0]] = -3.4e38f; }
    __syncthreads();
  }

  if (tid < TOPK) w[tid] = __expf((selv[tid] - selv[0]) * 25.0f); // 1/0.04
  __syncthreads();
  if (tid == 0) { float s = 0.f; for (int k = 0; k < TOPK; ++k) s += w[k]; wsum = s; }
  __syncthreads();

  if (tid < ES_N) {
    const int col = sp_idx[tid];
    float acc = 0.f;
    for (int k = 0; k < TOPK; ++k)
      acc += w[k] * es_db[(size_t)seli[k] * DIM + col];
    acc *= (1.0f / wsum);
    es_ret [(size_t)b * ES_N + tid] = acc;
    out_ret[(size_t)b * ES_N + tid] = acc;   // output 1: es_retrieved_101
  }
}

// ---------------------------------------------------------------------------
// Kernel 4: MLP in f32 WMMA 16x16x4. Block = 16 batch rows, 8 waves.
// feat (1125, zero-padded to 1128) staged in LDS; LN + exact GELU between
// layers; layer3 (N=101) uses 7 col-tiles with store guards.
// ---------------------------------------------------------------------------
__global__ void __launch_bounds__(256) mlp_kernel(
    const float* __restrict__ en, const float* __restrict__ es_ret,
    const float* __restrict__ W1, const float* __restrict__ b1,
    const float* __restrict__ g1, const float* __restrict__ be1,
    const float* __restrict__ W2, const float* __restrict__ b2,
    const float* __restrict__ g2, const float* __restrict__ be2,
    const float* __restrict__ W3, const float* __restrict__ b3,
    float* __restrict__ out_pred, float* __restrict__ out_delta)
{
  __shared__ float feat [16 * FK];    // ~72 KB
  __shared__ float h1buf[16 * H1_N];  // 16 KB
  __shared__ float h2buf[16 * H2_N];  // 8 KB

  const int tid   = threadIdx.x, wave = tid >> 5, lane = tid & 31;
  const int b0    = blockIdx.x * 16;
  const int mrow  = lane & 15;
  const int khalf = (lane < 16) ? 0 : 2;  // f32 A/B frag: K pair per lane half
  const int rbase = (lane < 16) ? 0 : 8;  // C frag: rows r / r+8

  for (int idx = tid; idx < 16 * FK; idx += 256) {
    int r = idx / FK, k = idx - r * FK;
    float v = 0.f;
    if (k < DIM)             v = en    [(size_t)(b0 + r) * DIM  + k];
    else if (k < DIM + ES_N) v = es_ret[(size_t)(b0 + r) * ES_N + (k - DIM)];
    feat[idx] = v;
  }
  __syncthreads();

  // ---- layer 1: 1125 -> 256 ----
  #pragma unroll
  for (int tt = 0; tt < 2; ++tt) {
    const int ncol = (wave + tt * 8) * 16 + (lane & 15);
    v8f c = {};
    for (int kc = 0; kc < FK; kc += 4) {
      int ka = kc + khalf;
      v2f a = { feat[mrow * FK + ka], feat[mrow * FK + ka + 1] };
      v2f bb = { (ka     < DIM + ES_N) ? W1[(size_t)ka       * H1_N + ncol] : 0.f,
                 (ka + 1 < DIM + ES_N) ? W1[(size_t)(ka + 1) * H1_N + ncol] : 0.f };
      c = __builtin_amdgcn_wmma_f32_16x16x4_f32(
              false, a, false, bb, (short)0, c, false, false);
    }
    float bias = b1[ncol];
    #pragma unroll
    for (int r = 0; r < 8; ++r) h1buf[(rbase + r) * H1_N + ncol] = c[r] + bias;
  }
  __syncthreads();

  // ---- LN + GELU over 256 (2 rows per wave) ----
  #pragma unroll
  for (int rr = 0; rr < 2; ++rr) {
    const int row = wave + rr * 8;
    float s = 0.f, s2 = 0.f, xv[8];
    #pragma unroll
    for (int i = 0; i < 8; ++i) {
      xv[i] = h1buf[row * H1_N + lane + 32 * i]; s += xv[i]; s2 += xv[i] * xv[i];
    }
    #pragma unroll
    for (int m = 16; m >= 1; m >>= 1) { s += __shfl_xor(s, m, 32); s2 += __shfl_xor(s2, m, 32); }
    float mu = s / H1_N, var = s2 / H1_N - mu * mu, rstd = rsqrtf(var + 1e-5f);
    #pragma unroll
    for (int i = 0; i < 8; ++i) {
      int col = lane + 32 * i;
      float y = (xv[i] - mu) * rstd * g1[col] + be1[col];
      h1buf[row * H1_N + col] = 0.5f * y * (1.f + erff(y * 0.70710678f));
    }
  }
  __syncthreads();

  // ---- layer 2: 256 -> 128 ----
  {
    const int ncol = wave * 16 + (lane & 15);
    v8f c = {};
    for (int kc = 0; kc < H1_N; kc += 4) {
      int ka = kc + khalf;
      v2f a = { h1buf[mrow * H1_N + ka], h1buf[mrow * H1_N + ka + 1] };
      v2f bb = { W2[(size_t)ka * H2_N + ncol], W2[(size_t)(ka + 1) * H2_N + ncol] };
      c = __builtin_amdgcn_wmma_f32_16x16x4_f32(
              false, a, false, bb, (short)0, c, false, false);
    }
    float bias = b2[ncol];
    #pragma unroll
    for (int r = 0; r < 8; ++r) h2buf[(rbase + r) * H2_N + ncol] = c[r] + bias;
  }
  __syncthreads();

  // ---- LN + GELU over 128 ----
  #pragma unroll
  for (int rr = 0; rr < 2; ++rr) {
    const int row = wave + rr * 8;
    float s = 0.f, s2 = 0.f, xv[4];
    #pragma unroll
    for (int i = 0; i < 4; ++i) {
      xv[i] = h2buf[row * H2_N + lane + 32 * i]; s += xv[i]; s2 += xv[i] * xv[i];
    }
    #pragma unroll
    for (int m = 16; m >= 1; m >>= 1) { s += __shfl_xor(s, m, 32); s2 += __shfl_xor(s2, m, 32); }
    float mu = s / H2_N, var = s2 / H2_N - mu * mu, rstd = rsqrtf(var + 1e-5f);
    #pragma unroll
    for (int i = 0; i < 4; ++i) {
      int col = lane + 32 * i;
      float y = (xv[i] - mu) * rstd * g2[col] + be2[col];
      h2buf[row * H2_N + col] = 0.5f * y * (1.f + erff(y * 0.70710678f));
    }
  }
  __syncthreads();

  // ---- layer 3: 128 -> 101 (7 tiles, last partial) ----
  if (wave < 7) {
    const int nraw = wave * 16 + (lane & 15);
    const int ncol = (nraw < ES_N) ? nraw : (ES_N - 1);  // clamp loads
    v8f c = {};
    for (int kc = 0; kc < H2_N; kc += 4) {
      int ka = kc + khalf;
      v2f a = { h2buf[mrow * H2_N + ka], h2buf[mrow * H2_N + ka + 1] };
      v2f bb = { W3[(size_t)ka * ES_N + ncol], W3[(size_t)(ka + 1) * ES_N + ncol] };
      c = __builtin_amdgcn_wmma_f32_16x16x4_f32(
              false, a, false, bb, (short)0, c, false, false);
    }
    if (nraw < ES_N) {
      float bias = b3[nraw];
      #pragma unroll
      for (int r = 0; r < 8; ++r) {
        int brow = b0 + rbase + r;
        float delta = c[r] + bias;
        float er = es_ret[(size_t)brow * ES_N + nraw];
        out_delta[(size_t)brow * ES_N + nraw] = delta;       // output 2
        out_pred [(size_t)brow * ES_N + nraw] = er + delta;  // output 0
      }
    }
  }
}

// ---------------------------------------------------------------------------
extern "C" void kernel_launch(void* const* d_in, const int* in_sizes, int n_in,
                              void* d_out, int out_size, void* d_ws, size_t ws_size,
                              hipStream_t stream)
{
  const float* en    = (const float*)d_in[0];
  const float* en_db = (const float*)d_in[1];
  const float* es_db = (const float*)d_in[2];
  const int*   sp    = (const int*)  d_in[3];
  const float* W1    = (const float*)d_in[4];
  const float* b1    = (const float*)d_in[5];
  const float* g1    = (const float*)d_in[6];
  const float* be1   = (const float*)d_in[7];
  const float* W2    = (const float*)d_in[8];
  const float* b2    = (const float*)d_in[9];
  const float* g2    = (const float*)d_in[10];
  const float* be2   = (const float*)d_in[11];
  const float* W3    = (const float*)d_in[12];
  const float* b3    = (const float*)d_in[13];
  float* out = (float*)d_out;

  char* ws = (char*)d_ws;
  size_t off = 0;
  auto carve = [&](size_t bytes) -> void* {
    void* p = ws + off;
    off = (off + bytes + 255) & ~(size_t)255;
    return p;
  };
  __bf16* qn   = (__bf16*)carve((size_t)B_ROWS * DIM * 2);
  __bf16* dbn  = (__bf16*)carve((size_t)N_DB  * DIM * 2);
  float* pvals = (float*) carve((size_t)B_ROWS * NSPLIT * TOPK * 4);
  int*   pidx  = (int*)   carve((size_t)B_ROWS * NSPLIT * TOPK * 4);
  float* esr   = (float*) carve((size_t)B_ROWS * ES_N * 4);

  hipLaunchKernelGGL(l2norm_to_bf16, dim3(B_ROWS / 8), dim3(256), 0, stream,
                     en, qn, B_ROWS);
  hipLaunchKernelGGL(l2norm_to_bf16, dim3(N_DB / 8), dim3(256), 0, stream,
                     en_db, dbn, N_DB);
  hipLaunchKernelGGL(sims_topk_kernel, dim3(B_ROWS / MROWS, NSPLIT), dim3(256), 0, stream,
                     qn, dbn, pvals, pidx);
  hipLaunchKernelGGL(merge_softmax_retrieve, dim3(B_ROWS), dim3(128), 0, stream,
                     pvals, pidx, es_db, sp, esr, out + (size_t)B_ROWS * ES_N);
  hipLaunchKernelGGL(mlp_kernel, dim3(B_ROWS / 16), dim3(256), 0, stream,
                     en, esr, W1, b1, g1, be1, W2, b2, g2, be2, W3, b3,
                     out, out + 2 * (size_t)B_ROWS * ES_N);
}
